// PeriodicKernelAttention_82343112999023
// MI455X (gfx1250) — matrix operand order repaired and checked by reading, not compiled, not run on hardware
//
#include <hip/hip_runtime.h>
#include <hip/hip_bf16.h>

// ---------------------------------------------------------------------------
// Periodic-kernel attention, flash-attention style, WMMA f16 (gfx1250).
// B=1, H=12, S=2048, D=64. sqrt(D)=8.
// score = -2*sin(pi*sqrt(max(2-2*cos,0)))^2/8 + (|q|^2+|k|^2)/16
// ---------------------------------------------------------------------------

constexpr int H = 12;
constexpr int S = 2048;
constexpr int D = 64;
constexpr float PI_F = 3.14159265358979323846f;

#define NEG_INF (-__builtin_inff())

typedef __attribute__((ext_vector_type(16))) _Float16 v16h;
typedef __attribute__((ext_vector_type(8)))  _Float16 half8;
typedef __attribute__((ext_vector_type(8)))  float    v8f;

union V16U { v16h v; half8 h[2]; };

// ---------------------------------------------------------------------------
// Prep 1: per-row L2 norms. One wave (32 lanes) per row, 2 floats per lane.
// Writes qn/kn (f16, row-major) and qsq/ksq (f32 squared norms).
// ---------------------------------------------------------------------------
__global__ __launch_bounds__(256) void prep_norm(
    const float* __restrict__ q, const float* __restrict__ k,
    _Float16* __restrict__ qn, _Float16* __restrict__ kn,
    float* __restrict__ qsq, float* __restrict__ ksq) {
  const int w    = threadIdx.x >> 5;
  const int lane = threadIdx.x & 31;
  const int row  = blockIdx.x * 8 + w;            // 0 .. H*S-1

  {
    const float2* qr = (const float2*)(q + (size_t)row * D);
    float2 vq = qr[lane];
    float ss = vq.x * vq.x + vq.y * vq.y;
#pragma unroll
    for (int m = 16; m >= 1; m >>= 1) ss += __shfl_xor(ss, m);
    float inv = rsqrtf(ss);
    qn[(size_t)row * D + 2 * lane]     = (_Float16)(vq.x * inv);
    qn[(size_t)row * D + 2 * lane + 1] = (_Float16)(vq.y * inv);
    if (lane == 0) qsq[row] = ss;
  }
  {
    const float2* kr = (const float2*)(k + (size_t)row * D);
    float2 vk = kr[lane];
    float ss = vk.x * vk.x + vk.y * vk.y;
#pragma unroll
    for (int m = 16; m >= 1; m >>= 1) ss += __shfl_xor(ss, m);
    float inv = rsqrtf(ss);
    kn[(size_t)row * D + 2 * lane]     = (_Float16)(vk.x * inv);
    kn[(size_t)row * D + 2 * lane + 1] = (_Float16)(vk.y * inv);
    if (lane == 0) ksq[row] = ss;
  }
}

// ---------------------------------------------------------------------------
// Prep 2: V -> f16, transposed per head: vt[h][d][s] = (f16)V[h][s][d]
// so P*V B-fragments are contiguous 16B loads along the key dimension.
// ---------------------------------------------------------------------------
__global__ __launch_bounds__(256) void prep_vt(
    const float* __restrict__ v, _Float16* __restrict__ vt) {
  int idx = blockIdx.x * 256 + threadIdx.x;       // over H*S*D
  int d  = idx & (D - 1);
  int hs = idx >> 6;
  int s  = hs & (S - 1);
  int h  = hs >> 11;
  vt[((size_t)h * D + d) * S + s] = (_Float16)v[idx];
}

// ---------------------------------------------------------------------------
// Flash attention. grid = (S/64, H); 128 threads = 4 waves; each wave owns a
// 16-row query tile. Per 32-key tile: 4 WMMA (QK^T, K-dim=64 chained) +
// fused periodic-kernel score + online softmax + LDS P-repack + 4 WMMA (P*V).
// ---------------------------------------------------------------------------
__global__ __launch_bounds__(128) void attn_kernel(
    const _Float16* __restrict__ qn, const _Float16* __restrict__ kn,
    const _Float16* __restrict__ vt, const float* __restrict__ qsq,
    const float* __restrict__ ksq, const int* __restrict__ mask,
    float* __restrict__ out) {
  __shared__ __align__(16) _Float16 pbuf[4][16][32];   // per-wave P repack

  const int tid   = threadIdx.x;
  const int w     = tid >> 5;
  const int lane  = tid & 31;
  const int lo    = lane & 15;
  const int hh    = lane >> 4;
  const int hid   = blockIdx.y;
  const int qrow0 = blockIdx.x * 64 + w * 16;

  // --- Q A-fragments (16x32 f16 layout): lanes 0-15 hold K 8h+0..7 / 16+8h+0..7
  const _Float16* qrow = qn + ((size_t)hid * S + qrow0 + lo) * D;
  V16U a0, a1;
  a0.h[0] = *(const half8*)(qrow + 8 * hh);
  a0.h[1] = *(const half8*)(qrow + 16 + 8 * hh);
  a1.h[0] = *(const half8*)(qrow + 32 + 8 * hh);
  a1.h[1] = *(const half8*)(qrow + 48 + 8 * hh);

  float qs[8];
#pragma unroll
  for (int r = 0; r < 8; ++r)
    qs[r] = qsq[(size_t)hid * S + qrow0 + r + 8 * hh];

  float m_run[8], l_run[8];
  v8f o[4];
  v8f zero8 = {};
#pragma unroll
  for (int r = 0; r < 8; ++r) { m_run[r] = NEG_INF; l_run[r] = 0.0f; }
#pragma unroll
  for (int nb = 0; nb < 4; ++nb) o[nb] = zero8;

  for (int kb = 0; kb < S; kb += 32) {
    if (kb + 32 < S)
      __builtin_prefetch(kn + ((size_t)hid * S + kb + 32) * D, 0, 1);

    float st0[8], st1[8];
    // --- two 16-key subtiles: QK^T + fused periodic-kernel score
#pragma unroll
    for (int sub = 0; sub < 2; ++sub) {
      const int nb0 = kb + 16 * sub;
      const _Float16* krow = kn + ((size_t)hid * S + nb0 + lo) * D;
      V16U b0u, b1u;
      b0u.h[0] = *(const half8*)(krow + 8 * hh);        // features 0..31
      b0u.h[1] = *(const half8*)(krow + 16 + 8 * hh);
      b1u.h[0] = *(const half8*)(krow + 32 + 8 * hh);   // features 32..63
      b1u.h[1] = *(const half8*)(krow + 48 + 8 * hh);
      v16h b0 = b0u.v, b1 = b1u.v;
      v8f c = __builtin_amdgcn_wmma_f32_16x16x32_f16(
          false, a0.v, false, b0, (short)0, zero8, false, false);
      c = __builtin_amdgcn_wmma_f32_16x16x32_f16(
          false, a1.v, false, b1, (short)0, c, false, false);

      float kq = ksq[(size_t)hid * S + nb0 + lo];
      float* st = sub ? st1 : st0;
#pragma unroll
      for (int r = 0; r < 8; ++r) {
        float cosv = c[r];
        float t  = fmaxf(2.0f - 2.0f * cosv, 0.0f);
        float sn = __sinf(PI_F * sqrtf(t));
        float sc = fmaf(sn * sn, -0.25f, (qs[r] + kq) * 0.0625f);
        int mrow = qrow0 + r + 8 * hh;
        if (mask[(size_t)mrow * S + nb0 + lo] == 0) sc = NEG_INF;
        st[r] = sc;
      }
    }

    // --- online softmax (row = r + 8*hh, 16-lane half reductions)
    float p0[8], p1[8];
#pragma unroll
    for (int r = 0; r < 8; ++r) {
      float mx = fmaxf(st0[r], st1[r]);
#pragma unroll
      for (int off = 8; off >= 1; off >>= 1) mx = fmaxf(mx, __shfl_xor(mx, off));
      float mnew = fmaxf(m_run[r], mx);
      float scale, e0, e1;
      if (mnew == NEG_INF) { scale = 1.0f; e0 = 0.0f; e1 = 0.0f; }
      else {
        scale = (m_run[r] == NEG_INF) ? 0.0f : __expf(m_run[r] - mnew);
        e0 = __expf(st0[r] - mnew);
        e1 = __expf(st1[r] - mnew);
      }
      float rs = e0 + e1;
#pragma unroll
      for (int off = 8; off >= 1; off >>= 1) rs += __shfl_xor(rs, off);
      l_run[r] = l_run[r] * scale + rs;
      m_run[r] = mnew;
#pragma unroll
      for (int nb = 0; nb < 4; ++nb) o[nb][r] *= scale;
      p0[r] = e0; p1[r] = e1;
    }

    // --- repack P (C layout -> A-fragment layout) through per-wave LDS.
    // Per-wave DS ops are in-order; compiler inserts s_wait_dscnt.
#pragma unroll
    for (int r = 0; r < 8; ++r) {
      pbuf[w][r + 8 * hh][lo]      = (_Float16)p0[r];
      pbuf[w][r + 8 * hh][16 + lo] = (_Float16)p1[r];
    }
    V16U pf;
    pf.h[0] = *(const half8*)&pbuf[w][lo][8 * hh];
    pf.h[1] = *(const half8*)&pbuf[w][lo][16 + 8 * hh];

    // --- O += P * V  (B-fragments from transposed V: contiguous along keys)
#pragma unroll
    for (int nb = 0; nb < 4; ++nb) {
      const _Float16* vrow =
          vt + ((size_t)hid * D + 16 * nb + lo) * S + kb + 8 * hh;
      V16U bvu;
      bvu.h[0] = *(const half8*)vrow;
      bvu.h[1] = *(const half8*)(vrow + 16);
      v16h bv = bvu.v;
      o[nb] = __builtin_amdgcn_wmma_f32_16x16x32_f16(
          false, pf.v, false, bv, (short)0, o[nb], false, false);
    }
  }

  // --- epilogue: normalize and store
  float invl[8];
#pragma unroll
  for (int r = 0; r < 8; ++r) invl[r] = 1.0f / l_run[r];
#pragma unroll
  for (int nb = 0; nb < 4; ++nb) {
#pragma unroll
    for (int r = 0; r < 8; ++r) {
      int row = qrow0 + r + 8 * hh;
      out[((size_t)hid * S + row) * D + 16 * nb + lo] = o[nb][r] * invl[r];
    }
  }
}

// ---------------------------------------------------------------------------
extern "C" void kernel_launch(void* const* d_in, const int* in_sizes, int n_in,
                              void* d_out, int out_size, void* d_ws, size_t ws_size,
                              hipStream_t stream) {
  const float* q    = (const float*)d_in[0];
  const float* k    = (const float*)d_in[1];
  const float* v    = (const float*)d_in[2];
  const int*   mask = (const int*)d_in[3];
  float*       out  = (float*)d_out;

  // Workspace layout (~9.2 MB):
  //   qn (f16 H*S*D) | kn (f16 H*S*D) | vt (f16 H*D*S) | qsq (f32 H*S) | ksq
  char* ws = (char*)d_ws;
  _Float16* qn = (_Float16*)ws;
  _Float16* kn = qn + (size_t)H * S * D;
  _Float16* vt = kn + (size_t)H * S * D;
  float* qsq = (float*)(vt + (size_t)H * S * D);
  float* ksq = qsq + (size_t)H * S;

  prep_norm<<<(H * S) / 8, 256, 0, stream>>>(q, k, qn, kn, qsq, ksq);
  prep_vt<<<(H * S * D) / 256, 256, 0, stream>>>(v, vt);

  dim3 grid(S / 64, H);
  attn_kernel<<<grid, 128, 0, stream>>>(qn, kn, vt, qsq, ksq, mask, out);
}
